// Head_36919538877068
// MI455X (gfx1250) — matrix operand order repaired and checked by reading, not compiled, not run on hardware
//
#include <hip/hip_runtime.h>

typedef __attribute__((ext_vector_type(16))) _Float16 v16h;
typedef __attribute__((ext_vector_type(8)))  _Float16 v8h;
typedef __attribute__((ext_vector_type(8)))  float    v8f;

#define WMMA_F16(A, B, C) \
  __builtin_amdgcn_wmma_f32_16x16x32_f16(false, (A), false, (B), (short)0, (C), false, false)

static constexpr int BATCH = 4;
static constexpr int T     = 2048;
static constexpr int CEMB  = 1024;
static constexpr int H     = 64;

static __device__ __forceinline__ v16h cat8(v8h lo, v8h hi) {
  return __builtin_shufflevector(lo, hi, 0,1,2,3,4,5,6,7,8,9,10,11,12,13,14,15);
}
static __device__ __forceinline__ unsigned lds_byte_off(const void* p) {
  return (unsigned)(size_t)p;   // low 32 bits of a flat LDS pointer = LDS byte address
}
// CDNA5 async copy: 16B per lane, global -> LDS, tracked by ASYNCcnt
static __device__ __forceinline__ void async_ld16(unsigned lds, const void* g) {
  asm volatile("global_load_async_to_lds_b128 %0, %1, off" :: "v"(lds), "v"(g) : "memory");
}

// ---------------------------------------------------------------------------
// Weight transpose + f32->f16 (+scale): W[1024x64] -> Wt[64x1024]
// ---------------------------------------------------------------------------
__global__ __launch_bounds__(256) void wtrans_kernel(const float* __restrict__ W,
                                                     _Float16* __restrict__ Wt,
                                                     float scale) {
  const int t   = blockIdx.x * 256 + threadIdx.x;   // 65536 threads
  const int c   = t & (CEMB - 1);
  const int col = t >> 10;
  Wt[(size_t)col * CEMB + c] = (_Float16)(W[(size_t)c * H + col] * scale);
}

// ---------------------------------------------------------------------------
// Projection: 8 waves/block, wave does a 16x64 tile. A from X (f32, vector
// loads + cvt), B from pre-transposed Wt (two 16B global reads per lane, L2-hot).
// K/Q written row-major f16; V written transposed (Vt[b][64][2048]).
// ---------------------------------------------------------------------------
__global__ __launch_bounds__(256) void proj_kernel(const float* __restrict__ X,
                                                   const _Float16* __restrict__ Wt,
                                                   _Float16* __restrict__ Yrow,
                                                   _Float16* __restrict__ Vt) {
  const int lane = threadIdx.x & 31;
  const int wave = threadIdx.x >> 5;
  const int hi   = (lane >= 16) ? 1 : 0;
  const int m    = lane & 15;
  const int r0   = (blockIdx.x * 8 + wave) * 16;
  const int b0   = hi ? 8 : 0;     // A-layout K-run starts
  const int b1   = hi ? 24 : 16;

  v8f acc[4] = {};
  for (int c = 0; c < CEMB; c += 32) {
    const float* xrow = X + (size_t)(r0 + m) * CEMB + c;
    float4 f0 = *(const float4*)(xrow + b0);
    float4 f1 = *(const float4*)(xrow + b0 + 4);
    float4 f2 = *(const float4*)(xrow + b1);
    float4 f3 = *(const float4*)(xrow + b1 + 4);
    if (c + 32 < CEMB) __builtin_prefetch(xrow + 32, 0, 1);
    v16h a;
    a[0]=(_Float16)f0.x;  a[1]=(_Float16)f0.y;  a[2]=(_Float16)f0.z;  a[3]=(_Float16)f0.w;
    a[4]=(_Float16)f1.x;  a[5]=(_Float16)f1.y;  a[6]=(_Float16)f1.z;  a[7]=(_Float16)f1.w;
    a[8]=(_Float16)f2.x;  a[9]=(_Float16)f2.y;  a[10]=(_Float16)f2.z; a[11]=(_Float16)f2.w;
    a[12]=(_Float16)f3.x; a[13]=(_Float16)f3.y; a[14]=(_Float16)f3.z; a[15]=(_Float16)f3.w;

#pragma unroll
    for (int nt = 0; nt < 4; ++nt) {
      const v16h b = cat8(*(const v8h*)(Wt + (size_t)(nt * 16 + m) * CEMB + c + b0),
                          *(const v8h*)(Wt + (size_t)(nt * 16 + m) * CEMB + c + b1));
      acc[nt] = WMMA_F16(a, b, acc[nt]);
    }
  }

  if (Vt != nullptr) {
    // store transposed: Vt[b][col][t]
#pragma unroll
    for (int nt = 0; nt < 4; ++nt)
#pragma unroll
      for (int r = 0; r < 8; ++r) {
        const int row = r0 + r + (hi ? 8 : 0);
        const int bb  = row >> 11;
        const int tt  = row & (T - 1);
        Vt[(size_t)bb * H * T + (size_t)(nt * 16 + m) * T + tt] = (_Float16)acc[nt][r];
      }
  } else {
#pragma unroll
    for (int nt = 0; nt < 4; ++nt)
#pragma unroll
      for (int r = 0; r < 8; ++r)
        Yrow[(size_t)(r0 + r + (hi ? 8 : 0)) * H + nt * 16 + m] = (_Float16)acc[nt][r];
  }
}

// ---------------------------------------------------------------------------
// Flash attention: one wave per (batch, 16-query tile), 32-wide key blocks.
// K blocks double-buffered into LDS with global_load_async_to_lds_b128;
// V read directly from transposed Vt (contiguous, L2-hot); P re-laid out
// C->A through per-wave LDS. Row sums computed with a WMMA against an
// all-ones B matrix (replaces 32 shuffles + 32 adds per block).
// ---------------------------------------------------------------------------
__global__ __launch_bounds__(128) void attn_kernel(const _Float16* __restrict__ Km,
                                                   const _Float16* __restrict__ Qm,
                                                   const _Float16* __restrict__ Vt,
                                                   float* __restrict__ Out) {
  __shared__ __align__(32) _Float16 lp[4][16 * 32];        // P tiles, 1KB/wave
  __shared__ __align__(32) _Float16 lk[4][2][32 * 64];     // K double-buffer, 8KB/wave

  const int lane = threadIdx.x & 31;
  const int wave = threadIdx.x >> 5;
  const int w    = blockIdx.x * 4 + wave;
  const int b    = w >> 7;                 // / (T/16)
  const int qt   = 127 - (w & 127);        // longest (most key blocks) tiles first
  const int q0   = qt * 16;
  const int m    = lane & 15;
  const int hi   = (lane >= 16) ? 1 : 0;
  const int b0   = hi ? 8 : 0;
  const int b1   = hi ? 24 : 16;
  const size_t base = (size_t)b * T * H;

  // Q A-operands (two 32-wide H chunks); contiguous 16B runs
  v16h qa[2];
  {
    const _Float16* qrow = Qm + base + (size_t)(q0 + m) * H;
#pragma unroll
    for (int hc = 0; hc < 2; ++hc)
      qa[hc] = cat8(*(const v8h*)(qrow + hc * 32 + b0),
                    *(const v8h*)(qrow + hc * 32 + b1));
  }

  // all-ones B operand for row-sum WMMA
  v16h vones;
#pragma unroll
  for (int i = 0; i < 16; ++i) vones[i] = (_Float16)1.0f;
  const v8f vzero = {};

  v8f o[4] = {};
  float mrun[8], lrun[8], alpha[8];
#pragma unroll
  for (int r = 0; r < 8; ++r) { mrun[r] = -__builtin_inff(); lrun[r] = 0.0f; }

  _Float16* myP = lp[wave];
  const _Float16* kb0p = lk[wave][0];
  const _Float16* kb1p = lk[wave][1];
  const unsigned  koff[2] = { lds_byte_off(kb0p), lds_byte_off(kb1p) };
  const char* Kbase = (const char*)(Km + base);
  const _Float16* vtb = Vt + (size_t)b * H * T;
  const int nblk = (qt >> 1) + 1;

  // async-stage key block kb into LDS buffer `buf` (32 rows x 128B = 4KB)
  auto prefetchK = [&](int kb, int buf) {
    const char* g = Kbase + (size_t)kb * 32 * 128 + lane * 16;
    const unsigned l = koff[buf] + lane * 16;
#pragma unroll
    for (int j = 0; j < 8; ++j) async_ld16(l + j * 512, g + j * 512);
  };

  prefetchK(0, 0);
  int cur = 0;

  for (int kb = 0; kb < nblk; ++kb) {
    const int k0 = kb * 32;
    asm volatile("s_wait_asynccnt 0" ::: "memory");         // current K buffer ready
    if (kb + 1 < nblk) prefetchK(kb + 1, cur ^ 1);          // overlap next block
    const _Float16* kbuf = cur ? kb1p : kb0p;

    // S = Q K^T : two 16-wide key n-tiles, contraction over H=64
    v8f s[2];
#pragma unroll
    for (int nt = 0; nt < 2; ++nt) {
      v8f cacc = {};
#pragma unroll
      for (int hc = 0; hc < 2; ++hc) {
        const v16h bm = cat8(*(const v8h*)(kbuf + (nt * 16 + m) * 64 + hc * 32 + b0),
                             *(const v8h*)(kbuf + (nt * 16 + m) * 64 + hc * 32 + b1));
        cacc = WMMA_F16(qa[hc], bm, cacc);
      }
      s[nt] = cacc;
    }

    // causal mask + online softmax (max via lane butterfly; sums via WMMA below)
#pragma unroll
    for (int r = 0; r < 8; ++r) {
      const int query = q0 + r + (hi ? 8 : 0);
      float s0 = (k0 + m      <= query) ? s[0][r] : -__builtin_inff();
      float s1 = (k0 + 16 + m <= query) ? s[1][r] : -__builtin_inff();
      float t = fmaxf(s0, s1);
#pragma unroll
      for (int msk = 1; msk < 16; msk <<= 1) t = fmaxf(t, __shfl_xor(t, msk, 32));
      const float mnew = fmaxf(mrun[r], t);
      alpha[r] = __expf(mrun[r] - mnew);
      mrun[r]  = mnew;
      const float p0 = __expf(s0 - mnew);
      const float p1 = __expf(s1 - mnew);
#pragma unroll
      for (int nt = 0; nt < 4; ++nt) o[nt][r] *= alpha[r];
      const int prow = r + (hi ? 8 : 0);
      myP[prow * 32 + m]      = (_Float16)p0;
      myP[prow * 32 + 16 + m] = (_Float16)p1;
    }
    asm volatile("s_wait_dscnt 0" ::: "memory");   // P staging visible to the wave

    // P as A operand: one contiguous 32B LDS read per lane (k = 16h + s)
    const v16h pa = *(const v16h*)(myP + m * 32 + (hi ? 16 : 0));

    // row sums via matrix unit: rsd[r] = sum_k P[row][k] (same value in every lane)
    const v8f rsd = WMMA_F16(pa, vones, vzero);
#pragma unroll
    for (int r = 0; r < 8; ++r) lrun[r] = lrun[r] * alpha[r] + rsd[r];

    // O += P @ V : B operand contiguous from transposed Vt (global, L2-hot)
#pragma unroll
    for (int nt = 0; nt < 4; ++nt) {
      const v16h bm = *(const v16h*)(vtb + (size_t)(nt * 16 + m) * T + k0 + (hi ? 16 : 0));
      o[nt] = WMMA_F16(pa, bm, o[nt]);
    }
    cur ^= 1;
  }

  // epilogue: normalize by row sum, write f32 output
#pragma unroll
  for (int nt = 0; nt < 4; ++nt)
#pragma unroll
    for (int r = 0; r < 8; ++r) {
      const int row = q0 + r + (hi ? 8 : 0);
      Out[base + (size_t)row * H + nt * 16 + m] = o[nt][r] / lrun[r];
    }
}

// ---------------------------------------------------------------------------
extern "C" void kernel_launch(void* const* d_in, const int* in_sizes, int n_in,
                              void* d_out, int out_size, void* d_ws, size_t ws_size,
                              hipStream_t stream) {
  (void)in_sizes; (void)n_in; (void)out_size; (void)ws_size;
  const float* x  = (const float*)d_in[0];
  const float* Wk = (const float*)d_in[1];
  const float* Wq = (const float*)d_in[2];
  const float* Wv = (const float*)d_in[3];

  _Float16* ws   = (_Float16*)d_ws;
  const size_t n  = (size_t)BATCH * T * H;   // 524288 halves per matrix
  const size_t wn = (size_t)H * CEMB;        // 65536 halves per weight
  _Float16* Kmat = ws;
  _Float16* Qmat = ws + n;
  _Float16* Vt   = ws + 2 * n;
  _Float16* WtK  = ws + 3 * n;
  _Float16* WtQ  = WtK + wn;
  _Float16* WtV  = WtQ + wn;

  // transpose + convert weights (Q gets c^-0.5 = 1/32 folded in)
  wtrans_kernel<<<dim3(256), 256, 0, stream>>>(Wk, WtK, 1.0f);
  wtrans_kernel<<<dim3(256), 256, 0, stream>>>(Wq, WtQ, 0.03125f);
  wtrans_kernel<<<dim3(256), 256, 0, stream>>>(Wv, WtV, 1.0f);

  // 8192 rows / (8 waves * 16 rows) = 64 blocks per projection
  proj_kernel<<<dim3(64), 256, 0, stream>>>(x, WtK, Kmat, nullptr);
  proj_kernel<<<dim3(64), 256, 0, stream>>>(x, WtQ, Qmat, nullptr);
  proj_kernel<<<dim3(64), 256, 0, stream>>>(x, WtV, nullptr, Vt);

  // 4 batches * 128 query tiles / 4 waves per block = 128 blocks
  attn_kernel<<<dim3(128), 128, 0, stream>>>(Kmat, Qmat, Vt, (float*)d_out);
}
